// NodeGenerator_56831007260864
// MI455X (gfx1250) — matrix operand order repaired and checked by reading, not compiled, not run on hardware
//
#include <hip/hip_runtime.h>
#include <hip/hip_bf16.h>
#include <math.h>

#define BDIM 8
#define TDIM 1024
#define SDIM 1024
#define HDIM 1024
#define VDIM 5000
#define MROWS (BDIM * TDIM) /* 8192 */

typedef __attribute__((ext_vector_type(16))) __bf16 v16bf;
typedef __attribute__((ext_vector_type(8)))  __bf16 bf16x8;
typedef __attribute__((ext_vector_type(8)))  float  v8f;

union Frag { v16bf v; bf16x8 h[2]; };

#define LDT 40 /* padded LDS row stride (elems): 80B rows, 16B aligned, conflict-free */

// Split f32 into bf16 hi (RNE) + bf16 lo residual: x ~= hi + lo.
static __device__ __forceinline__ void split_f32_bf16(float x, __bf16& hi, __bf16& lo) {
    unsigned u = __float_as_uint(x);
    unsigned r = u + 0x7FFFu + ((u >> 16) & 1u);
    unsigned hb = r & 0xFFFF0000u;
    float hf = __uint_as_float(hb);
    hi = __builtin_bit_cast(__bf16, (unsigned short)(hb >> 16));
    lo = (__bf16)(x - hf);
}

static __device__ __forceinline__ float gelu_tanh(float x) {
    const float c0 = 0.7978845608028654f; // sqrt(2/pi)
    const float c1 = 0.044715f;
    float t = tanhf(c0 * (x + c1 * x * x * x));
    return 0.5f * x * (1.0f + t);
}

// Shared per-K-step fragment load + 12x WMMA (bf16x3 split emulation) for one
// wave computing a 2x2 grid of 16x16 tiles out of a 128x64 block tile.
static __device__ __forceinline__ void compute_step(const __bf16 (*__restrict__ Ah)[LDT],
                                                    const __bf16 (*__restrict__ Al)[LDT],
                                                    const __bf16 (*__restrict__ Bh)[LDT],
                                                    const __bf16 (*__restrict__ Bl)[LDT],
                                                    int wm0, int wn0, int lm, int hf,
                                                    v8f acc[2][2]) {
    Frag ah[2], al[2], bh[2], bl[2];
#pragma unroll
    for (int i = 0; i < 2; ++i) {
        int row = wm0 + i * 16 + lm;
        ah[i].h[0] = *(const bf16x8*)&Ah[row][hf * 8];
        ah[i].h[1] = *(const bf16x8*)&Ah[row][16 + hf * 8];
        al[i].h[0] = *(const bf16x8*)&Al[row][hf * 8];
        al[i].h[1] = *(const bf16x8*)&Al[row][16 + hf * 8];
    }
#pragma unroll
    for (int j = 0; j < 2; ++j) {
        int col = wn0 + j * 16 + lm;
        bh[j].h[0] = *(const bf16x8*)&Bh[col][hf * 8];
        bh[j].h[1] = *(const bf16x8*)&Bh[col][16 + hf * 8];
        bl[j].h[0] = *(const bf16x8*)&Bl[col][hf * 8];
        bl[j].h[1] = *(const bf16x8*)&Bl[col][16 + hf * 8];
    }
#pragma unroll
    for (int i = 0; i < 2; ++i)
#pragma unroll
        for (int j = 0; j < 2; ++j) {
            acc[i][j] = __builtin_amdgcn_wmma_f32_16x16x32_bf16(
                false, ah[i].v, false, bh[j].v, (short)0, acc[i][j], false, false);
            acc[i][j] = __builtin_amdgcn_wmma_f32_16x16x32_bf16(
                false, al[i].v, false, bh[j].v, (short)0, acc[i][j], false, false);
            acc[i][j] = __builtin_amdgcn_wmma_f32_16x16x32_bf16(
                false, ah[i].v, false, bl[j].v, (short)0, acc[i][j], false, false);
        }
}

// ---------------------------------------------------------------------------
// Kernel 1: p_gen[row] = sigmoid(x[row,:] . Wg + bg)
// ---------------------------------------------------------------------------
__global__ __launch_bounds__(256) void pgen_kernel(const float* __restrict__ x,
                                                   const float* __restrict__ Wg,
                                                   const float* __restrict__ bg,
                                                   float* __restrict__ p_gen) {
    __shared__ float red[256];
    const int row = blockIdx.x;
    float s = 0.f;
    for (int k = threadIdx.x; k < HDIM; k += 256)
        s += x[(size_t)row * HDIM + k] * Wg[k];
    red[threadIdx.x] = s;
    __syncthreads();
    for (int off = 128; off > 0; off >>= 1) {
        if (threadIdx.x < off) red[threadIdx.x] += red[threadIdx.x + off];
        __syncthreads();
    }
    if (threadIdx.x == 0) {
        float z = red[0] + bg[0];
        p_gen[row] = 1.f / (1.f + expf(-z));
    }
}

// ---------------------------------------------------------------------------
// Kernel 2: h = gelu(x @ Wf + bf), stored as bf16 hi/lo pair (split precision)
// Double-buffered LDS ping-pong: next tile's global loads issue before the
// current tile's WMMAs; LDS stores land in the other buffer; 1 barrier/iter.
// ---------------------------------------------------------------------------
__global__ __launch_bounds__(256) void gemm1_fc_gelu(const float* __restrict__ x,
                                                     const float* __restrict__ Wf,
                                                     const float* __restrict__ bfb,
                                                     __bf16* __restrict__ h_hi,
                                                     __bf16* __restrict__ h_lo) {
    __shared__ __bf16 Ah[2][128][LDT];
    __shared__ __bf16 Al[2][128][LDT];
    __shared__ __bf16 Bh[2][64][LDT];  // transposed: Bh[buf][n][k]
    __shared__ __bf16 Bl[2][64][LDT];

    const int tid  = threadIdx.x;
    const int lane = tid & 31;
    const int wave = tid >> 5;
    const int lm   = lane & 15;
    const int hf   = lane >> 4;
    const int wm0  = (wave & 3) * 32;
    const int wn0  = (wave >> 2) * 32;
    const int bm0  = blockIdx.x * 128;
    const int bn0  = blockIdx.y * 64;

    v8f acc[2][2] = {};

    float4 aReg[4]; // 128x32 f32 tile staged: 4 x float4 per thread
    float  bReg[8]; // 32x64 f32 tile staged: 8 floats per thread

    auto load_stage = [&](int k0) {
#pragma unroll
        for (int it = 0; it < 4; ++it) {
            int i = tid + it * 256;
            int m = i >> 3, k = (i & 7) * 4;
            aReg[it] = *(const float4*)&x[(size_t)(bm0 + m) * HDIM + k0 + k];
        }
#pragma unroll
        for (int it = 0; it < 8; ++it) {
            int i = tid + it * 256;
            int k = i >> 6, n = i & 63;
            bReg[it] = Wf[(size_t)(k0 + k) * HDIM + bn0 + n];
        }
    };
    auto store_stage = [&](int buf) {
#pragma unroll
        for (int it = 0; it < 4; ++it) {
            int i = tid + it * 256;
            int m = i >> 3, k = (i & 7) * 4;
            split_f32_bf16(aReg[it].x, Ah[buf][m][k + 0], Al[buf][m][k + 0]);
            split_f32_bf16(aReg[it].y, Ah[buf][m][k + 1], Al[buf][m][k + 1]);
            split_f32_bf16(aReg[it].z, Ah[buf][m][k + 2], Al[buf][m][k + 2]);
            split_f32_bf16(aReg[it].w, Ah[buf][m][k + 3], Al[buf][m][k + 3]);
        }
#pragma unroll
        for (int it = 0; it < 8; ++it) {
            int i = tid + it * 256;
            int k = i >> 6, n = i & 63;
            split_f32_bf16(bReg[it], Bh[buf][n][k], Bl[buf][n][k]);
        }
    };

    load_stage(0);
    store_stage(0);
    __syncthreads();

    const int NT = HDIM / 32;
    for (int kt = 0; kt < NT; ++kt) {
        const int cur = kt & 1;
        const bool has_next = (kt + 1) < NT;
        if (has_next) load_stage((kt + 1) * 32);
        compute_step(Ah[cur], Al[cur], Bh[cur], Bl[cur], wm0, wn0, lm, hf, acc);
        if (has_next) store_stage(cur ^ 1);
        __syncthreads();
    }

    // Epilogue: bias + gelu, store h as split bf16 hi/lo
#pragma unroll
    for (int i = 0; i < 2; ++i)
#pragma unroll
        for (int j = 0; j < 2; ++j) {
            int gn = bn0 + wn0 + j * 16 + lm;
            float bias = bfb[gn];
#pragma unroll
            for (int r = 0; r < 8; ++r) {
                int gm = bm0 + wm0 + i * 16 + hf * 8 + r;
                float g = gelu_tanh(acc[i][j][r] + bias);
                __bf16 hi, lo;
                split_f32_bf16(g, hi, lo);
                size_t off = (size_t)gm * HDIM + gn;
                h_hi[off] = hi;
                h_lo[off] = lo;
            }
        }
}

// ---------------------------------------------------------------------------
// Kernel 3: logits = h @ Wv + bv -> d_out (N = 5000, guarded tail tile),
// same double-buffered structure; A comes pre-split from workspace.
// ---------------------------------------------------------------------------
__global__ __launch_bounds__(256) void gemm2_vocab(const __bf16* __restrict__ h_hi,
                                                   const __bf16* __restrict__ h_lo,
                                                   const float* __restrict__ Wv,
                                                   const float* __restrict__ bv,
                                                   float* __restrict__ out) {
    __shared__ __bf16 Ah[2][128][LDT];
    __shared__ __bf16 Al[2][128][LDT];
    __shared__ __bf16 Bh[2][64][LDT];
    __shared__ __bf16 Bl[2][64][LDT];

    const int tid  = threadIdx.x;
    const int lane = tid & 31;
    const int wave = tid >> 5;
    const int lm   = lane & 15;
    const int hf   = lane >> 4;
    const int wm0  = (wave & 3) * 32;
    const int wn0  = (wave >> 2) * 32;
    const int bm0  = blockIdx.x * 128;
    const int bn0  = blockIdx.y * 64;

    v8f acc[2][2] = {};

    unsigned aRegH[8], aRegL[8]; // 128x32 bf16 tiles: 8 dwords (2 bf16 each) per thread
    float    bReg[8];

    auto load_stage = [&](int k0) {
#pragma unroll
        for (int it = 0; it < 8; ++it) {
            int i = tid + it * 256;
            int m = i >> 4, k = (i & 15) * 2;
            size_t off = (size_t)(bm0 + m) * HDIM + k0 + k;
            aRegH[it] = *(const unsigned*)&h_hi[off];
            aRegL[it] = *(const unsigned*)&h_lo[off];
        }
#pragma unroll
        for (int it = 0; it < 8; ++it) {
            int i = tid + it * 256;
            int k = i >> 6, n = i & 63;
            int gn = bn0 + n;
            if (gn >= VDIM) gn = VDIM - 1; // clamped load; never stored
            bReg[it] = Wv[(size_t)(k0 + k) * VDIM + gn];
        }
    };
    auto store_stage = [&](int buf) {
#pragma unroll
        for (int it = 0; it < 8; ++it) {
            int i = tid + it * 256;
            int m = i >> 4, k = (i & 15) * 2;
            *(unsigned*)&Ah[buf][m][k] = aRegH[it];
            *(unsigned*)&Al[buf][m][k] = aRegL[it];
        }
#pragma unroll
        for (int it = 0; it < 8; ++it) {
            int i = tid + it * 256;
            int k = i >> 6, n = i & 63;
            split_f32_bf16(bReg[it], Bh[buf][n][k], Bl[buf][n][k]);
        }
    };

    load_stage(0);
    store_stage(0);
    __syncthreads();

    const int NT = HDIM / 32;
    for (int kt = 0; kt < NT; ++kt) {
        const int cur = kt & 1;
        const bool has_next = (kt + 1) < NT;
        if (has_next) load_stage((kt + 1) * 32);
        compute_step(Ah[cur], Al[cur], Bh[cur], Bl[cur], wm0, wn0, lm, hf, acc);
        if (has_next) store_stage(cur ^ 1);
        __syncthreads();
    }

#pragma unroll
    for (int i = 0; i < 2; ++i)
#pragma unroll
        for (int j = 0; j < 2; ++j) {
            int gn = bn0 + wn0 + j * 16 + lm;
            if (gn < VDIM) {
                float bias = bv[gn];
#pragma unroll
                for (int r = 0; r < 8; ++r) {
                    int gm = bm0 + wm0 + i * 16 + hf * 8 + r;
                    out[(size_t)gm * VDIM + gn] = acc[i][j][r] + bias;
                }
            }
        }
}

// ---------------------------------------------------------------------------
// Kernel 4: per-row softmax * p_gen + LDS-histogram pointer scatter
// ---------------------------------------------------------------------------
__global__ __launch_bounds__(256) void finalize_kernel(float* __restrict__ out,
                                                       const float* __restrict__ p_gen,
                                                       const float* __restrict__ attn,
                                                       const int* __restrict__ nodes) {
    __shared__ float srow[VDIM]; // 20 KB logits/exp row
    __shared__ float hist[VDIM]; // 20 KB vocab scatter accumulator
    __shared__ float red[256];
    const int tid = threadIdx.x;
    const int row = blockIdx.x;       // row = b*T + t
    const int b   = row / TDIM;
    float* orow = out + (size_t)row * VDIM;

    float lmax = -3.402823466e38f;
    for (int v = tid; v < VDIM; v += 256) {
        float lv = orow[v];
        srow[v] = lv;
        hist[v] = 0.f;
        lmax = fmaxf(lmax, lv);
    }
    red[tid] = lmax;
    __syncthreads();
    for (int off = 128; off > 0; off >>= 1) {
        if (tid < off) red[tid] = fmaxf(red[tid], red[tid + off]);
        __syncthreads();
    }
    float mx = red[0];
    __syncthreads();

    float lsum = 0.f;
    for (int v = tid; v < VDIM; v += 256) {
        float e = expf(srow[v] - mx);
        srow[v] = e;
        lsum += e;
    }
    red[tid] = lsum;
    __syncthreads();
    for (int off = 128; off > 0; off >>= 1) {
        if (tid < off) red[tid] += red[tid + off];
        __syncthreads();
    }
    float p     = p_gen[row];
    float scale = p / red[0];
    float q     = 1.f - p;

    // pointer/copy scatter via LDS float atomics (ds_add_f32)
    for (int s = tid; s < SDIM; s += 256) {
        int idx = nodes[(size_t)b * SDIM + s];
        float w = attn[(size_t)row * SDIM + s] * q;
        atomicAdd(&hist[idx], w);
    }
    __syncthreads();

    for (int v = tid; v < VDIM; v += 256)
        orow[v] = srow[v] * scale + hist[v];
}

// ---------------------------------------------------------------------------
extern "C" void kernel_launch(void* const* d_in, const int* in_sizes, int n_in,
                              void* d_out, int out_size, void* d_ws, size_t ws_size,
                              hipStream_t stream) {
    const float* x    = (const float*)d_in[0];
    const float* attn = (const float*)d_in[1];
    const int*   nodes= (const int*)  d_in[2];
    const float* Wg   = (const float*)d_in[3];
    const float* bg   = (const float*)d_in[4];
    const float* Wf   = (const float*)d_in[5];
    const float* bfb  = (const float*)d_in[6];
    const float* Wv   = (const float*)d_in[7];
    const float* bv   = (const float*)d_in[8];
    float* out = (float*)d_out;

    // Workspace layout: p_gen (M f32, 256B-aligned) | h_hi (M*H bf16) | h_lo (M*H bf16)
    size_t pgen_bytes = (((size_t)MROWS * sizeof(float)) + 255) & ~(size_t)255;
    float*  p_gen = (float*)d_ws;
    __bf16* h_hi  = (__bf16*)((char*)d_ws + pgen_bytes);
    __bf16* h_lo  = h_hi + (size_t)MROWS * HDIM;

    pgen_kernel<<<MROWS, 256, 0, stream>>>(x, Wg, bg, p_gen);
    gemm1_fc_gelu<<<dim3(MROWS / 128, HDIM / 64), 256, 0, stream>>>(x, Wf, bfb, h_hi, h_lo);
    gemm2_vocab<<<dim3(MROWS / 128, (VDIM + 63) / 64), 256, 0, stream>>>(h_hi, h_lo, Wv, bv, out);
    finalize_kernel<<<MROWS, 256, 0, stream>>>(out, p_gen, attn, nodes);
}
